// LinearAttention_73254962201207
// MI455X (gfx1250) — matrix operand order repaired and checked by reading, not compiled, not run on hardware
//
#include <hip/hip_runtime.h>

typedef __bf16 bf16_t;
typedef __attribute__((ext_vector_type(16))) __bf16 v16bf;
typedef __attribute__((ext_vector_type(8)))  __bf16 v8bf;
typedef __attribute__((ext_vector_type(8)))  float  v8f;

#define B_  2
#define N_  2048
#define C_  1024
#define H_  16
#define D_  64
#define BH_ (B_ * H_)          // 32
#define M_  (B_ * N_)          // 4096
#define KSPLIT 8               // split-K chunks for the KV GEMM (2048/8 = 256)

// ---- WMMA tile loaders (CDNA5 16x16x32 bf16 layouts, ISA 7.12.2) ----

// A: 16x32 (MxK), row-major source, ld in elements.
// lane<16:  elems 0..7 = K(k0+0..7),  8..15 = K(k0+16..23) of row (lane&15)
// lane>=16: elems 0..7 = K(k0+8..15), 8..15 = K(k0+24..31)
__device__ __forceinline__ v16bf load_a16x32(const bf16_t* __restrict__ src,
                                             int ld, int lane, int k0) {
  const int m  = lane & 15;
  const int kb = k0 + ((lane >> 4) << 3);
  const bf16_t* p = src + m * ld + kb;
  v8bf lo = *(const v8bf*)(p);
  v8bf hi = *(const v8bf*)(p + 16);
  v16bf r;
#pragma unroll
  for (int i = 0; i < 8; ++i) { r[i] = lo[i]; r[i + 8] = hi[i]; }
  return r;
}

// B: 32x16 (KxN). srcT rows are B columns (srcT[col][k]), ld in elements.
// lane<16: column (lane&15), elems 0..7 = K(k0+0..7), 8..15 = K(k0+16..23); lane>=16: K(k0+8..15), K(k0+24..31)
__device__ __forceinline__ v16bf load_b32x16(const bf16_t* __restrict__ srcT,
                                             int ld, int lane, int k0) {
  const int c  = lane & 15;
  const int kb = k0 + ((lane >> 4) << 3);
  const bf16_t* p = srcT + c * ld + kb;
  v8bf lo = *(const v8bf*)(p);
  v8bf hi = *(const v8bf*)(p + 16);
  v16bf r;
#pragma unroll
  for (int i = 0; i < 8; ++i) { r[i] = lo[i]; r[i + 8] = hi[i]; }
  return r;
}

__device__ __forceinline__ v8f wmma_bf16(v16bf a, v16bf b, v8f c) {
  return __builtin_amdgcn_wmma_f32_16x16x32_bf16(false, a, false, b,
                                                 (short)0, c, false, false);
}

__device__ __forceinline__ float elu1(float x) {
  return x > 0.f ? x + 1.f : __expf(x);   // elu(x)+1
}

// ---- 1. f32 -> bf16 conversions ----

__global__ void k_cvt_x(const float* __restrict__ x, bf16_t* __restrict__ xb, int n) {
  int i = blockIdx.x * blockDim.x + threadIdx.x;
  if (i < n) xb[i] = (bf16_t)x[i];
}

// Wb rows: [0,1024)=Wq, [1024,3072)=Wkv, [3072,4096)=Wo; all [row][1024]
__global__ void k_cvt_w(const float* __restrict__ Wq, const float* __restrict__ Wkv,
                        const float* __restrict__ Wo, bf16_t* __restrict__ Wb) {
  int i = blockIdx.x * blockDim.x + threadIdx.x;   // 0 .. 4096*1024-1
  int row = i >> 10;
  float v;
  if (row < 1024)      v = Wq[i];
  else if (row < 3072) v = Wkv[i - (1024 << 10)];
  else                 v = Wo[i - (3072 << 10)];
  Wb[i] = (bf16_t)v;
}

// ---- 2. fused projection GEMM: [4096 x 3072 x 1024], 32x64 per wave ----
// epilogue scatters to q[bh][n][d] (elu+1), kT[bh][d][n] (elu+1), vT[bh][d][n]
__global__ void k_proj(const bf16_t* __restrict__ xb, const bf16_t* __restrict__ Wb,
                       bf16_t* __restrict__ qb, bf16_t* __restrict__ kT,
                       bf16_t* __restrict__ vT) {
  const int lane = threadIdx.x & 31;
  const int wave = blockIdx.x * (blockDim.x >> 5) + (threadIdx.x >> 5);
  const int mt = wave / 48;                 // 0..127  (32-row block)
  const int ns = wave % 48;                 // 0..47   (64-col strip of 3072)
  const bf16_t* A0 = xb + (size_t)mt * 32 * C_;
  const bf16_t* A1 = A0 + 16 * C_;
  v8f acc[2][4] = {};
  for (int k0 = 0; k0 < C_; k0 += 32) {
    v16bf a0 = load_a16x32(A0, C_, lane, k0);
    v16bf a1 = load_a16x32(A1, C_, lane, k0);
#pragma unroll
    for (int t = 0; t < 4; ++t) {
      v16bf b = load_b32x16(Wb + (size_t)(ns * 64 + t * 16) * C_, C_, lane, k0);
      acc[0][t] = wmma_bf16(a0, b, acc[0][t]);
      acc[1][t] = wmma_bf16(a1, b, acc[1][t]);
    }
  }
  const int cbase = ns * 64 + (lane & 15);
#pragma unroll
  for (int u = 0; u < 2; ++u) {
    const int rbase = mt * 32 + u * 16 + ((lane >> 4) << 3);
#pragma unroll
    for (int t = 0; t < 4; ++t) {
      const int col = cbase + t * 16;       // 0..3071, uniform tile per wave
#pragma unroll
      for (int v = 0; v < 8; ++v) {
        const int row = rbase + v;          // 0..4095
        const int bb = row >> 11, n = row & (N_ - 1);
        float val = acc[u][t][v];
        if (col < C_) {                     // q path
          int h = col >> 6, d = col & 63;
          qb[((bb * H_ + h) * N_ + n) * D_ + d] = (bf16_t)elu1(val);
        } else if (col < 2 * C_) {          // k path (store transposed)
          int jc = col - C_; int h = jc >> 6, d = jc & 63;
          kT[((bb * H_ + h) * D_ + d) * N_ + n] = (bf16_t)elu1(val);
        } else {                            // v path (store transposed)
          int jc = col - 2 * C_; int h = jc >> 6, d = jc & 63;
          vT[((bb * H_ + h) * D_ + d) * N_ + n] = (bf16_t)val;
        }
      }
    }
  }
}

// ---- 3. per-head KV = k^T v : [64 x 64 x 2048], split-K, f32 partials ----
// KVp[kc][bh][d2][d1] (transposed store so stage-4 B-loads are contiguous)
__global__ void k_kv(const bf16_t* __restrict__ kT, const bf16_t* __restrict__ vT,
                     float* __restrict__ KVp) {
  const int lane = threadIdx.x & 31;
  const int wave = blockIdx.x * (blockDim.x >> 5) + (threadIdx.x >> 5);
  const int kc = wave & (KSPLIT - 1);       // K chunk
  const int rt = (wave >> 3) & 1;           // 32-row strip of d1
  const int bh = wave >> 4;                 // 0..31
  const bf16_t* A0 = kT + (size_t)(bh * D_ + rt * 32) * N_;
  const bf16_t* A1 = A0 + 16 * N_;
  const bf16_t* Bb = vT + (size_t)bh * D_ * N_;
  const int kbeg = kc * (N_ / KSPLIT), kend = kbeg + (N_ / KSPLIT);
  v8f acc[2][4] = {};
  for (int k0 = kbeg; k0 < kend; k0 += 32) {
    v16bf a0 = load_a16x32(A0, N_, lane, k0);
    v16bf a1 = load_a16x32(A1, N_, lane, k0);
#pragma unroll
    for (int t = 0; t < 4; ++t) {
      v16bf b = load_b32x16(Bb + (size_t)t * 16 * N_, N_, lane, k0);
      acc[0][t] = wmma_bf16(a0, b, acc[0][t]);
      acc[1][t] = wmma_bf16(a1, b, acc[1][t]);
    }
  }
  float* dst = KVp + (size_t)(kc * BH_ + bh) * D_ * D_;
  const int d2 = lane & 15;
#pragma unroll
  for (int u = 0; u < 2; ++u) {
    const int d1 = rt * 32 + u * 16 + ((lane >> 4) << 3);
#pragma unroll
    for (int t = 0; t < 4; ++t)
#pragma unroll
      for (int v = 0; v < 8; ++v)
        dst[(t * 16 + d2) * D_ + (d1 + v)] = acc[u][t][v];
  }
}

// ---- 3r. reduce split-K partials -> bf16 KVt[bh][d2][d1] ----
__global__ void k_kv_reduce(const float* __restrict__ KVp, bf16_t* __restrict__ KVt) {
  int i = blockIdx.x * blockDim.x + threadIdx.x;   // 0 .. 32*64*64-1
  float s = 0.f;
#pragma unroll
  for (int kc = 0; kc < KSPLIT; ++kc) s += KVp[(size_t)kc * BH_ * D_ * D_ + i];
  KVt[i] = (bf16_t)s;
}

// ---- 3b. ksum[bh][d] = sum_n k[bh][n][d]  (one wave per row of kT) ----
__global__ void k_ksum(const bf16_t* __restrict__ kT, float* __restrict__ ksum) {
  const int lane = threadIdx.x & 31;
  const int row = blockIdx.x * (blockDim.x >> 5) + (threadIdx.x >> 5); // bh*64+d
  const bf16_t* p = kT + (size_t)row * N_;
  float s = 0.f;
  for (int n = lane; n < N_; n += 32) s += (float)p[n];
#pragma unroll
  for (int off = 16; off > 0; off >>= 1) s += __shfl_xor(s, off, 32);
  if (lane == 0) ksum[row] = s;
}

// ---- 3c. Z[bh][n] = 1 / (q[n] . ksum + 1e-6) ----
__global__ void k_z(const bf16_t* __restrict__ qb, const float* __restrict__ ksum,
                    float* __restrict__ Z) {
  int i = blockIdx.x * blockDim.x + threadIdx.x;   // bh*N + n
  int bh = i >> 11;
  const bf16_t* qrow = qb + (size_t)i * D_;
  const float* ks = ksum + bh * D_;
  float s = 0.f;
#pragma unroll
  for (int d = 0; d < D_; ++d) s += (float)qrow[d] * ks[d];
  Z[i] = 1.f / (s + 1e-6f);
}

// ---- 4. out_h = Z * (q @ KV) : per head [2048 x 64 x 64], 32x64 per wave ----
__global__ void k_attn(const bf16_t* __restrict__ qb, const bf16_t* __restrict__ KVt,
                       const float* __restrict__ Z, bf16_t* __restrict__ attn) {
  const int lane = threadIdx.x & 31;
  const int wave = blockIdx.x * (blockDim.x >> 5) + (threadIdx.x >> 5);
  const int bh = wave >> 6;                 // 0..31
  const int ms = wave & 63;                 // 32-row strip of N
  const bf16_t* A0 = qb + (size_t)(bh * N_ + ms * 32) * D_;
  const bf16_t* A1 = A0 + 16 * D_;
  const bf16_t* Bb = KVt + (size_t)bh * D_ * D_;
  v8f acc[2][4] = {};
#pragma unroll
  for (int k0 = 0; k0 < D_; k0 += 32) {
    v16bf a0 = load_a16x32(A0, D_, lane, k0);
    v16bf a1 = load_a16x32(A1, D_, lane, k0);
#pragma unroll
    for (int t = 0; t < 4; ++t) {
      v16bf b = load_b32x16(Bb + t * 16 * D_, D_, lane, k0);
      acc[0][t] = wmma_bf16(a0, b, acc[0][t]);
      acc[1][t] = wmma_bf16(a1, b, acc[1][t]);
    }
  }
  const int bb = bh >> 4, h = bh & 15;
  const int d2 = lane & 15;
#pragma unroll
  for (int u = 0; u < 2; ++u) {
    const int nb = ms * 32 + u * 16 + ((lane >> 4) << 3);
#pragma unroll
    for (int t = 0; t < 4; ++t)
#pragma unroll
      for (int v = 0; v < 8; ++v) {
        int n = nb + v;
        float val = acc[u][t][v] * Z[bh * N_ + n];
        attn[((size_t)(bb * N_ + n)) * C_ + h * D_ + t * 16 + d2] = (bf16_t)val;
      }
  }
}

// ---- 5. final GEMM: out = attn @ Wo^T + bo : [4096 x 1024 x 1024], f32 out ----
__global__ void k_out(const bf16_t* __restrict__ attn, const bf16_t* __restrict__ Wb,
                      const float* __restrict__ bo, float* __restrict__ out) {
  const int lane = threadIdx.x & 31;
  const int wave = blockIdx.x * (blockDim.x >> 5) + (threadIdx.x >> 5);
  const int mt = wave >> 4;                 // 0..127 (32-row block)
  const int ns = wave & 15;                 // 64-col strip of 1024
  const bf16_t* A0 = attn + (size_t)mt * 32 * C_;
  const bf16_t* A1 = A0 + 16 * C_;
  const bf16_t* Bb = Wb + (size_t)(3072 + ns * 64) * C_;   // Wo rows
  v8f acc[2][4] = {};
  for (int k0 = 0; k0 < C_; k0 += 32) {
    v16bf a0 = load_a16x32(A0, C_, lane, k0);
    v16bf a1 = load_a16x32(A1, C_, lane, k0);
#pragma unroll
    for (int t = 0; t < 4; ++t) {
      v16bf b = load_b32x16(Bb + (size_t)t * 16 * C_, C_, lane, k0);
      acc[0][t] = wmma_bf16(a0, b, acc[0][t]);
      acc[1][t] = wmma_bf16(a1, b, acc[1][t]);
    }
  }
  const int cb = ns * 64 + (lane & 15);
#pragma unroll
  for (int u = 0; u < 2; ++u) {
    const int rb = mt * 32 + u * 16 + ((lane >> 4) << 3);
#pragma unroll
    for (int t = 0; t < 4; ++t) {
      const int col = cb + t * 16;
      const float bias = bo[col];
#pragma unroll
      for (int v = 0; v < 8; ++v)
        out[(size_t)(rb + v) * C_ + col] = acc[u][t][v] + bias;
    }
  }
}

extern "C" void kernel_launch(void* const* d_in, const int* in_sizes, int n_in,
                              void* d_out, int out_size, void* d_ws, size_t ws_size,
                              hipStream_t stream) {
  const float* x   = (const float*)d_in[0];
  const float* Wq  = (const float*)d_in[1];
  const float* Wkv = (const float*)d_in[2];
  const float* Wo  = (const float*)d_in[3];
  const float* bo  = (const float*)d_in[4];
  float* out = (float*)d_out;

  char* ws = (char*)d_ws;
  size_t off = 0;
  auto alloc = [&](size_t bytes) -> char* {
    char* p = ws + off;
    off += (bytes + 255) & ~(size_t)255;
    return p;
  };
  bf16_t* xb   = (bf16_t*)alloc((size_t)M_ * C_ * 2);        // x in bf16
  bf16_t* Wb   = (bf16_t*)alloc((size_t)4096 * C_ * 2);      // [Wq;Wkv;Wo] bf16
  bf16_t* qb   = (bf16_t*)alloc((size_t)BH_ * N_ * D_ * 2);  // elu(q)+1
  bf16_t* kT   = (bf16_t*)alloc((size_t)BH_ * D_ * N_ * 2);  // (elu(k)+1)^T
  bf16_t* vT   = (bf16_t*)alloc((size_t)BH_ * D_ * N_ * 2);  // v^T
  float*  KVp  = (float*) alloc((size_t)KSPLIT * BH_ * D_ * D_ * 4); // split-K partials
  bf16_t* KVt  = (bf16_t*)alloc((size_t)BH_ * D_ * D_ * 2);  // (k^T v)^T
  float*  ksum = (float*) alloc((size_t)BH_ * D_ * 4);
  float*  Zb   = (float*) alloc((size_t)BH_ * N_ * 4);
  bf16_t* attn = (bf16_t*)alloc((size_t)M_ * C_ * 2);        // pre-Wo output

  k_cvt_x<<<(M_ * C_) / 256, 256, 0, stream>>>(x, xb, M_ * C_);
  k_cvt_w<<<(4096 * C_) / 256, 256, 0, stream>>>(Wq, Wkv, Wo, Wb);
  k_proj<<<(128 * 48) / 4, 128, 0, stream>>>(xb, Wb, qb, kT, vT);
  k_kv  <<<(BH_ * 2 * KSPLIT) / 4, 128, 0, stream>>>(kT, vT, KVp);
  k_kv_reduce<<<(BH_ * D_ * D_) / 256, 256, 0, stream>>>(KVp, KVt);
  k_ksum<<<(BH_ * D_) / 4, 128, 0, stream>>>(kT, ksum);
  k_z   <<<(BH_ * N_) / 256, 256, 0, stream>>>(qb, ksum, Zb);
  k_attn<<<(BH_ * 64) / 4, 128, 0, stream>>>(qb, KVt, Zb, attn);
  k_out <<<(128 * 16) / 4, 128, 0, stream>>>(attn, Wb, bo, out);
}
